// AttentionLSTM_76682346103214
// MI455X (gfx1250) — matrix operand order repaired and checked by reading, not compiled, not run on hardware
//
#include <hip/hip_runtime.h>
#include <hip/hip_bf16.h>
#include <math.h>

// Problem dims (fixed by the reference)
#define N_  256
#define T_  32
#define D_  1024
#define H_  1024
#define K_  3072        // D + H + H  (concat GEMM depth)
#define J_  4096        // 4*H       (gate output width)
#define KC  128         // K elements per pipeline stage
#define NSTAGE (K_/KC)  // 24
#define LDP (KC + 8)    // padded LDS row pitch (breaks 64-dword bank stride)

typedef __bf16 bf16;
typedef __attribute__((ext_vector_type(16))) __bf16 v16bf;
typedef __attribute__((ext_vector_type(8)))  __bf16 v8bf;
typedef __attribute__((ext_vector_type(8)))  float  v8f;
typedef int v4i __attribute__((vector_size(16)));   // int4, matches builtin param

#if __has_builtin(__builtin_amdgcn_global_load_async_to_lds_b128)
#define HAVE_ASYNC_LDS 1
#else
#define HAVE_ASYNC_LDS 0
#endif

__device__ __forceinline__ float sigmoidf_(float x) { return 1.0f / (1.0f + __expf(-x)); }

// 16-byte global -> LDS copy; async (ASYNCcnt-tracked) when available.
__device__ __forceinline__ void cp_b128(const bf16* g, bf16* l) {
#if HAVE_ASYNC_LDS
    __builtin_amdgcn_global_load_async_to_lds_b128(
        (__attribute__((address_space(1))) v4i*)g,   // global src (AS1), const dropped
        (__attribute__((address_space(3))) v4i*)l,   // LDS dst (AS3)
        0 /*offset*/, 0 /*cpol*/);
#else
    *(v8bf*)l = *(const v8bf*)g;   // synchronous fallback
#endif
}

template <int Ncnt>
__device__ __forceinline__ void wait_async() {
#if HAVE_ASYNC_LDS
# if __has_builtin(__builtin_amdgcn_s_wait_asynccnt)
    __builtin_amdgcn_s_wait_asynccnt(Ncnt);
# else
    if constexpr (Ncnt == 0) asm volatile("s_wait_asynccnt 0x0" ::: "memory");
    else                     asm volatile("s_wait_asynccnt 0x2" ::: "memory");
# endif
#endif
}

// A tile 16x32 bf16: lane(row=lane&15) holds K range
// [half*16, half*16+16) (contiguous, matching B).
__device__ __forceinline__ v16bf load_a(const bf16* ap, int half) {
    v8bf lo = *(const v8bf*)(ap + half * 16);
    v8bf hi = *(const v8bf*)(ap + half * 16 + 8);
    v16bf a;
#pragma unroll
    for (int i = 0; i < 8; ++i) { a[i] = lo[i]; a[8 + i] = hi[i]; }
    return a;
}

// ---------------------------------------------------------------------------
// Phase 0a: WT = [Wx; Wh; Wattn]^T as bf16, shape (J_ rows, K_ cols).
// ---------------------------------------------------------------------------
__global__ void prep_wt(const float* __restrict__ Wx, const float* __restrict__ Wh,
                        const float* __restrict__ Wattn, bf16* __restrict__ WT) {
    int idx = blockIdx.x * blockDim.x + threadIdx.x;   // 0 .. J_*K_-1
    int j = idx & (J_ - 1);
    int k = idx >> 12;
    float v;
    if (k < D_)              v = Wx[(size_t)k * J_ + j];
    else if (k < D_ + H_)    v = Wh[(size_t)(k - D_) * J_ + j];
    else                     v = Wattn[(size_t)(k - D_ - H_) * J_ + j];
    WT[(size_t)j * K_ + k] = (bf16)v;
}

__global__ void prep_x(const float* __restrict__ x, bf16* __restrict__ Xbf) {
    int i = blockIdx.x * blockDim.x + threadIdx.x;
    Xbf[i] = (bf16)x[i];
}

__global__ void init_state(const float* __restrict__ A, float* __restrict__ Hf,
                           float* __restrict__ Cf, bf16* __restrict__ Hbf) {
    int i = blockIdx.x * blockDim.x + threadIdx.x;     // 0 .. N_*H_-1
    const float* a = A + (size_t)i * 16;
    float s = 0.f;
#pragma unroll
    for (int l = 0; l < 16; ++l) s += a[l];
    s *= (1.0f / 16.0f);
    Hf[i] = s;  Cf[i] = s;  Hbf[i] = (bf16)s;
}

// ---------------------------------------------------------------------------
// Per-step attention (one block per sample, deterministic LDS tree reduction).
// ---------------------------------------------------------------------------
__global__ void attn_kernel(const float* __restrict__ A, const float* __restrict__ Hf,
                            bf16* __restrict__ Attnbf) {
    __shared__ float sp[256 * 16];
    __shared__ float wsh[16];
    const int n   = blockIdx.x;
    const int tid = threadIdx.x;
    const float* An = A + (size_t)n * H_ * 16;
    const float* hn = Hf + (size_t)n * H_;

    float s[16];
#pragma unroll
    for (int l = 0; l < 16; ++l) s[l] = 0.f;
    for (int q = 0; q < 4; ++q) {
        int h = q * 256 + tid;
        float hv = hn[h];
        const float* ar = An + (size_t)h * 16;
#pragma unroll
        for (int l = 0; l < 16; ++l) s[l] += hv * ar[l];
    }
#pragma unroll
    for (int l = 0; l < 16; ++l) sp[tid * 16 + l] = s[l];
    __syncthreads();
    for (int off = 128; off > 0; off >>= 1) {
        if (tid < off) {
#pragma unroll
            for (int l = 0; l < 16; ++l) sp[tid * 16 + l] += sp[(tid + off) * 16 + l];
        }
        __syncthreads();
    }
    if (tid == 0) {
        const float scale = 0.03125f;   // 1/sqrt(1024)
        float m = -1e30f;
#pragma unroll
        for (int l = 0; l < 16; ++l) { float z = sp[l] * scale; m = z > m ? z : m; }
        float e[16], sum = 0.f;
#pragma unroll
        for (int l = 0; l < 16; ++l) { e[l] = __expf(sp[l] * scale - m); sum += e[l]; }
        float inv = 1.0f / sum;
#pragma unroll
        for (int l = 0; l < 16; ++l) wsh[l] = e[l] * inv;
    }
    __syncthreads();
    for (int q = 0; q < 4; ++q) {
        int h = q * 256 + tid;
        const float* ar = An + (size_t)h * 16;
        float acc = 0.f;
#pragma unroll
        for (int l = 0; l < 16; ++l) acc += ar[l] * wsh[l];
        Attnbf[(size_t)n * H_ + h] = (bf16)acc;
    }
}

// ---------------------------------------------------------------------------
// Fused cell step: preact = [xt|h|attn] @ [Wx;Wh;Wattn]^T + b, gates, c/h.
// Block = 512 threads = 16 waves; block owns one 16-col h-tile, wave w owns
// m-tile w. B tiles (4 gates) staged in LDS via async copy, double-buffered,
// reused by all 16 waves. 4 WMMAs per 32-deep k-step, 4 k-steps per stage.
// ---------------------------------------------------------------------------
__global__ void __launch_bounds__(512) cell_kernel(
        const bf16* __restrict__ Xbf, const bf16* __restrict__ HbfIn,
        const bf16* __restrict__ Attnbf, const bf16* __restrict__ WT,
        const float* __restrict__ bias, float* __restrict__ Cf,
        float* __restrict__ Hf, bf16* __restrict__ HbfOut,
        float* __restrict__ out, int t) {
    // B staging: [buf][gate*16+col][k] with padded pitch LDP. 2 x ~17 KB.
    __shared__ __attribute__((aligned(32))) bf16 Blds[2][64 * LDP];

    const int tid  = threadIdx.x;
    const int lane = tid & 31;
    const int half = lane >> 4;
    const int col  = lane & 15;
    const int w    = tid >> 5;                 // wave = m-tile, 0..15
    const int hbase = blockIdx.x * 16;         // 64 blocks cover H columns

    // ---- cooperative B copy addressing: 16 KB/stage, 32 B per thread ----
    const int rid  = tid >> 3;                 // 0..63 = gate*16 + bcol
    const int part = tid & 7;                  // 8 threads per 256B row
    const int g_   = rid >> 4;
    const int bcol = rid & 15;
    const bf16* gsrc = WT + (size_t)(g_ * H_ + hbase + bcol) * K_ + part * 16;
    bf16* ldst = &Blds[0][0] + rid * LDP + part * 16;

    // ---- A operand row pointers for this wave's m-tile ----
    const int rowm = 16 * w + col;
    const bf16* ax = Xbf    + ((size_t)rowm * T_ + t) * D_;   // k in [0,1024)
    const bf16* ah = HbfIn  + (size_t)rowm * H_;              // k in [1024,2048)
    const bf16* aa = Attnbf + (size_t)rowm * H_;              // k in [2048,3072)

    // ---- accumulators: 4 gates, init with bias ----
    v8f acc0, acc1, acc2, acc3;
    const float b0 = bias[0 * H_ + hbase + col];
    const float b1 = bias[1 * H_ + hbase + col];
    const float b2 = bias[2 * H_ + hbase + col];
    const float b3 = bias[3 * H_ + hbase + col];
#pragma unroll
    for (int r = 0; r < 8; ++r) { acc0[r] = b0; acc1[r] = b1; acc2[r] = b2; acc3[r] = b3; }

    // prologue: issue stage-0 copy into buffer 0
#pragma unroll
    for (int jj = 0; jj < 2; ++jj)
        cp_b128(gsrc + 8 * jj, ldst + 8 * jj);

    for (int s = 0; s < NSTAGE; ++s) {
        const int cur = s & 1;
        if (s + 1 < NSTAGE) {
            // issue next stage's copy into the other buffer, then retire cur
            const bf16* gn = gsrc + (size_t)(s + 1) * KC;
            bf16* ln = ldst + (cur ^ 1) * (64 * LDP);
#pragma unroll
            for (int jj = 0; jj < 2; ++jj)
                cp_b128(gn + 8 * jj, ln + 8 * jj);
            wait_async<2>();    // stage-s copies done; next stage in flight
        } else {
            wait_async<0>();
        }
        __syncthreads();

        // A source for this stage (uniform branch, hoisted out of k-steps)
        const bf16* ap;
        if (s < 8)       ap = ax + s * KC;
        else if (s < 16) ap = ah + (s - 8) * KC;
        else             ap = aa + (s - 16) * KC;

        const bf16* bb = &Blds[cur][0];
#pragma unroll
        for (int kk = 0; kk < KC / 32; ++kk) {
            const int kb = kk * 32;
            v16bf av = load_a(ap + kb, half);
            // B 32x16 bf16: lane holds column `col`, K range [half*16, +16)
            v16bf bv0 = *(const v16bf*)(bb + (0 * 16 + col) * LDP + kb + half * 16);
            v16bf bv1 = *(const v16bf*)(bb + (1 * 16 + col) * LDP + kb + half * 16);
            v16bf bv2 = *(const v16bf*)(bb + (2 * 16 + col) * LDP + kb + half * 16);
            v16bf bv3 = *(const v16bf*)(bb + (3 * 16 + col) * LDP + kb + half * 16);
            acc0 = __builtin_amdgcn_wmma_f32_16x16x32_bf16(false, av, false, bv0, (short)0, acc0, false, false);
            acc1 = __builtin_amdgcn_wmma_f32_16x16x32_bf16(false, av, false, bv1, (short)0, acc1, false, false);
            acc2 = __builtin_amdgcn_wmma_f32_16x16x32_bf16(false, av, false, bv2, (short)0, acc2, false, false);
            acc3 = __builtin_amdgcn_wmma_f32_16x16x32_bf16(false, av, false, bv3, (short)0, acc3, false, false);
        }
        __syncthreads();   // all waves done with buf[cur] before it is refilled
    }

    // ---- epilogue: gates in registers. C/D element = (M=r+8*half, N=col) ----
    const int mbase = 16 * w;
    const int hcol  = hbase + col;
#pragma unroll
    for (int r = 0; r < 8; ++r) {
        const int m = mbase + r + 8 * half;
        const size_t si = (size_t)m * H_ + hcol;
        float iv = sigmoidf_(acc0[r]);
        float fv = sigmoidf_(acc1[r]);
        float ov = sigmoidf_(acc2[r]);
        float gv = tanhf(acc3[r]);
        float cp = Cf[si];
        float cn = fv * cp + iv * gv;
        float hn = ov * tanhf(cn);
        Cf[si] = cn;
        Hf[si] = hn;
        HbfOut[si] = (bf16)hn;
        out[((size_t)m * T_ + t) * H_ + hcol] = hn;   // hs[n][t][:] = h_{t+1}
    }
}

// ---------------------------------------------------------------------------
extern "C" void kernel_launch(void* const* d_in, const int* in_sizes, int n_in,
                              void* d_out, int out_size, void* d_ws, size_t ws_size,
                              hipStream_t stream) {
    const float* x     = (const float*)d_in[0];
    const float* A     = (const float*)d_in[1];
    const float* Wx    = (const float*)d_in[2];
    const float* Wh    = (const float*)d_in[3];
    const float* Wattn = (const float*)d_in[4];
    const float* bias  = (const float*)d_in[5];
    float* out = (float*)d_out;

    char* p = (char*)d_ws;
    bf16*  WT     = (bf16*)p;  p += (size_t)J_ * K_ * sizeof(bf16);        // 24 MB
    bf16*  Xbf    = (bf16*)p;  p += (size_t)N_ * T_ * D_ * sizeof(bf16);   // 16 MB
    float* Hf     = (float*)p; p += (size_t)N_ * H_ * sizeof(float);       //  1 MB
    float* Cf     = (float*)p; p += (size_t)N_ * H_ * sizeof(float);       //  1 MB
    bf16*  Hbf0   = (bf16*)p;  p += (size_t)N_ * H_ * sizeof(bf16);        // .5 MB
    bf16*  Hbf1   = (bf16*)p;  p += (size_t)N_ * H_ * sizeof(bf16);        // .5 MB
    bf16*  Attnbf = (bf16*)p;  p += (size_t)N_ * H_ * sizeof(bf16);        // .5 MB

    prep_wt<<<(J_ * K_) / 256, 256, 0, stream>>>(Wx, Wh, Wattn, WT);
    prep_x<<<(N_ * T_ * D_) / 256, 256, 0, stream>>>(x, Xbf);
    init_state<<<(N_ * H_) / 256, 256, 0, stream>>>(A, Hf, Cf, Hbf0);

    for (int t = 0; t < T_; ++t) {
        bf16* hin  = (t & 1) ? Hbf1 : Hbf0;   // ping-pong h buffers
        bf16* hout = (t & 1) ? Hbf0 : Hbf1;
        attn_kernel<<<N_, 256, 0, stream>>>(A, Hf, Attnbf);
        cell_kernel<<<64, 512, 0, stream>>>(Xbf, hin, Attnbf, WT, bias,
                                            Cf, Hf, hout, out, t);
    }
}
